// GemmaAttention_63307817943540
// MI455X (gfx1250) — matrix operand order repaired and checked by reading, not compiled, not run on hardware
//
#include <hip/hip_runtime.h>
#include <hip/hip_bf16.h>
#include <math.h>

#define L_   4096
#define D_   2048
#define H_   16
#define KV_  4
#define HD_  128
#define WIN_ 1024
#define NEG_INF (-3.0e38f)

typedef __attribute__((ext_vector_type(16))) __bf16 v16bf;
typedef __attribute__((ext_vector_type(8)))  __bf16 v8bf;
typedef __attribute__((ext_vector_type(8)))  float  v8f;

union FragAB { v16bf v; v8bf h[2]; };

__device__ __forceinline__ void sched_fence() {
#if __has_builtin(__builtin_amdgcn_sched_barrier)
  __builtin_amdgcn_sched_barrier(0);
#endif
}

__device__ __forceinline__ void wait_asynccnt0() {
#if __has_builtin(__builtin_amdgcn_s_wait_asynccnt)
  __builtin_amdgcn_s_wait_asynccnt(0);
#else
  asm volatile("s_wait_asynccnt 0" ::: "memory");
#endif
}

// Async copy of 16 bytes from global memory to LDS (CDNA5 GLOBAL_LOAD_ASYNC_TO_LDS_B128).
// lds_addr: 32-bit LDS byte address (generic shared pointer truncated per ISA 10.2).
__device__ __forceinline__ void async_copy_b128(unsigned lds_addr, const void* gaddr) {
  asm volatile("global_load_async_to_lds_b128 %0, %1, off"
               :
               : "v"(lds_addr), "v"((unsigned long long)(size_t)gaddr)
               : "memory");
}

// ---------------------------------------------------------------- converts

__global__ void f32_to_bf16_kernel(const float* __restrict__ in,
                                   __bf16* __restrict__ out, int n) {
  int i = blockIdx.x * blockDim.x + threadIdx.x;
  if (i < n) out[i] = (__bf16)in[i];
}

// W[k][n] (rows x cols, fp32) -> WT[n][k] (bf16)
__global__ void transpose_f32_to_bf16_kernel(const float* __restrict__ W,
                                             __bf16* __restrict__ WT,
                                             int rows, int cols) {
  int i = blockIdx.x * blockDim.x + threadIdx.x;
  if (i >= rows * cols) return;
  int n = i / rows;
  int k = i - n * rows;
  WT[(size_t)n * rows + k] = (__bf16)W[(size_t)k * cols + n];
}

// v fp32 [L][KV][HD] -> vt bf16 [KV][HD][L]
__global__ void v_transpose_bf16_kernel(const float* __restrict__ v,
                                        __bf16* __restrict__ vt) {
  int i = blockIdx.x * blockDim.x + threadIdx.x;
  if (i >= L_ * KV_ * HD_) return;
  int l = i & (L_ - 1);
  int t = i >> 12;            // / L_
  int d = t & (HD_ - 1);
  int kv = t >> 7;
  vt[i] = (__bf16)v[((size_t)l * KV_ + kv) * HD_ + d];
}

// ---------------------------------------------------------------- GEMM
// C[M][N] (fp32) = A[M][K] (bf16, row-major) * BT[N][K]^T (bf16, row-major)
// Block = 4 waves = 64 M-rows x 64 N-cols. B panel (64x32 bf16 per k-step) is
// shared by all waves -> staged in LDS via async copy. Manual two-phase
// ping-pong (a0/Bsh[0] vs a1/Bsh[1]) avoids register copies in the loop.
template <int K>
__global__ __launch_bounds__(128) void gemm_bf16t_kernel(
    const __bf16* __restrict__ A,
    const __bf16* __restrict__ BT,
    float* __restrict__ C,
    int M, int N) {
  static_assert((K % 64) == 0, "K must be a multiple of 64");
  const int tid  = threadIdx.x;
  const int lane = tid & 31;
  const int wave = tid >> 5;
  const int m0 = (blockIdx.y * 4 + wave) * 16;
  const int n0 = blockIdx.x * 64;
  const int half = lane >> 4;
  const int l15 = lane & 15;

  __shared__ __bf16 Bsh[2][64 * 32];   // 2 x 4KB double buffer

  // Stage one 64x32 k-chunk of the B panel into Bsh[buf].
  // 256 x 16B segments; 128 threads copy 2 segments each (async -> LDS).
  auto stageB = [&](int buf, int kc) {
#pragma unroll
    for (int rep = 0; rep < 2; ++rep) {
      int seg = tid + rep * 128;
      int row = seg >> 2;
      int col8 = (seg & 3) * 8;
      const __bf16* gp = BT + (size_t)(n0 + row) * K + kc + col8;
      __bf16* lp = &Bsh[buf][row * 32 + col8];
      async_copy_b128((unsigned)(size_t)lp, gp);
    }
  };

  const __bf16* arow = A + (size_t)(m0 + l15) * K;
  const int aoff = half ? 8 : 0;    // A frag: lanes 0-15 K {0..7,16..23}, lanes 16-31 K {8..15,24..31}
  const int boff = half ? 8 : 0;    // B frag: lanes 0-15 K {0..7,16..23}, lanes 16-31 K {8..15,24..31}

  auto loadA = [&](FragAB& f, int kc) {
    f.h[0] = *(const v8bf*)(arow + kc + aoff);
    f.h[1] = *(const v8bf*)(arow + kc + aoff + 16);
  };
  // One k-chunk of WMMAs: 4 B fragments from LDS, then 4 WMMAs.
  v8f acc[4] = {};
  auto compute = [&](const FragAB& a, int buf) {
    FragAB b[4];
#pragma unroll
    for (int j = 0; j < 4; ++j) {
      const __bf16* lrow = &Bsh[buf][(j * 16 + l15) * 32];
      b[j].h[0] = *(const v8bf*)(lrow + boff);
      b[j].h[1] = *(const v8bf*)(lrow + boff + 16);
    }
    sched_fence();   // keep all 8 ds loads issued before the first WMMA
#pragma unroll
    for (int j = 0; j < 4; ++j)
      acc[j] = __builtin_amdgcn_wmma_f32_16x16x32_bf16(
          false, a.v, false, b[j].v, (short)0, acc[j], false, false);
  };

  FragAB a0, a1;
  loadA(a0, 0);
  stageB(0, 0);

  for (int kc = 0; kc < K; kc += 64) {
    // ---- phase 0: compute on a0 / Bsh[0]; prefetch kc+32 into a1 / Bsh[1]
    loadA(a1, kc + 32);                    // kc+32 <= K-32 always (K % 64 == 0)
    wait_asynccnt0();
    __syncthreads();                       // Bsh[0] ready for all waves
    stageB(1, kc + 32);
    compute(a0, 0);
    // ---- phase 1: compute on a1 / Bsh[1]; prefetch kc+64 into a0 / Bsh[0]
    const bool more = (kc + 64) < K;
    if (more) loadA(a0, kc + 64);
    wait_asynccnt0();
    __syncthreads();                       // Bsh[1] ready for all waves
    if (more) stageB(0, kc + 64);
    compute(a1, 1);
  }
#pragma unroll
  for (int j = 0; j < 4; ++j)
#pragma unroll
    for (int r = 0; r < 8; ++r)
      C[(size_t)(m0 + half * 8 + r) * N + (n0 + j * 16 + l15)] = acc[j][r];
}

// ---------------------------------------------------------------- RMSNorm + RoPE
// in: fp32 [L][NH][HD]; out: bf16 [NH][L][HD]
__global__ __launch_bounds__(128) void rmsnorm_rope_kernel(
    const float* __restrict__ in, const float* __restrict__ scale,
    const int* __restrict__ positions, __bf16* __restrict__ out, int NH) {
  const int l = blockIdx.x, h = blockIdx.y, d = threadIdx.x;
  __shared__ float red[128];
  __shared__ float nrm[128];
  float v = in[((size_t)l * NH + h) * HD_ + d];
  red[d] = v * v;
  __syncthreads();
  for (int s = 64; s > 0; s >>= 1) {
    if (d < s) red[d] += red[d + s];
    __syncthreads();
  }
  float r = rsqrtf(red[0] * (1.0f / HD_) + 1e-6f);
  nrm[d] = v * r * scale[d];
  __syncthreads();
  int hi = d & 63;
  float ts  = __powf(10000.0f, -(float)hi / 64.0f);
  float ang = (float)positions[l] * ts;
  float sn = __sinf(ang), cs = __cosf(ang);
  float x1 = nrm[hi], x2 = nrm[hi + 64];
  float o = (d < 64) ? (x1 * cs - x2 * sn) : (x2 * cs + x1 * sn);
  out[((size_t)h * L_ + l) * HD_ + d] = (__bf16)o;
}

// ---------------------------------------------------------------- Flash attention
// Q:[H][L][HD] bf16, K:[KV][L][HD] bf16, VT:[KV][HD][L] bf16 -> AO:[L][H*HD] bf16
// 1 wave per (16-query tile, head); key tiles of 32.
__global__ __launch_bounds__(128) void attn_kernel(
    const __bf16* __restrict__ Q,
    const __bf16* __restrict__ Kc,
    const __bf16* __restrict__ VT,
    __bf16* __restrict__ AO) {
  const int lane = threadIdx.x & 31;
  const int wave = threadIdx.x >> 5;
  const int qt = blockIdx.x * 4 + wave;
  const int h  = blockIdx.y;
  const int q0 = qt * 16;
  const int kv = h / (H_ / KV_);
  const int half = lane >> 4;
  const int l15  = lane & 15;

  __shared__ __bf16 plds[4][16 * 32];
  __bf16* P = plds[wave];

  // Q fragments: 4 chunks of K-dim 32 covering HD=128
  FragAB qf[4];
  {
    const __bf16* qrow = Q + ((size_t)h * L_ + (q0 + l15)) * HD_;
    const int off = half ? 8 : 0;
#pragma unroll
    for (int c = 0; c < 4; ++c) {
      qf[c].h[0] = *(const v8bf*)(qrow + 32 * c + off);
      qf[c].h[1] = *(const v8bf*)(qrow + 32 * c + off + 16);
    }
  }

  v8f o[8] = {};
  float mrow[8], lrow[8];
#pragma unroll
  for (int r = 0; r < 8; ++r) { mrow[r] = NEG_INF; lrow[r] = 0.0f; }

  const __bf16* Kb = Kc + (size_t)kv * L_ * HD_;
  const __bf16* Vb = VT + (size_t)kv * HD_ * L_;
  const float scl = 0.088388347648318447f;  // 1/sqrt(128)

  int kstart = q0 - (WIN_ - 1);
  if (kstart < 0) kstart = 0;
  kstart &= ~31;
  const int kend = q0 + 15;

  for (int kt = kstart; kt <= kend; kt += 32) {
    // ---- issue ALL fragment loads for this tile up front:
    //      K frags feed the score WMMAs; V frags are independent of softmax,
    //      so their latency hides under the softmax VALU/trans work.
    FragAB kf[2][4];
#pragma unroll
    for (int t = 0; t < 2; ++t) {
      int key = kt + t * 16 + l15;
      if (key > L_ - 1) key = L_ - 1;
      const __bf16* krow = Kb + (size_t)key * HD_;
      const int boff = half ? 8 : 0;
#pragma unroll
      for (int c = 0; c < 4; ++c) {
        kf[t][c].h[0] = *(const v8bf*)(krow + 32 * c + boff);
        kf[t][c].h[1] = *(const v8bf*)(krow + 32 * c + boff + 16);
      }
    }
    FragAB vfr[8];
#pragma unroll
    for (int d8 = 0; d8 < 8; ++d8) {
      const __bf16* vrow = Vb + (size_t)(d8 * 16 + l15) * L_ + kt;
      const int boff = half ? 8 : 0;
      vfr[d8].h[0] = *(const v8bf*)(vrow + boff);
      vfr[d8].h[1] = *(const v8bf*)(vrow + boff + 16);
    }
    sched_fence();   // all 32 vmem loads issued before the first score WMMA

    // ---- scores: two 16x16 tiles covering 32 keys
    v8f s[2] = {};
#pragma unroll
    for (int t = 0; t < 2; ++t)
#pragma unroll
      for (int c = 0; c < 4; ++c)
        s[t] = __builtin_amdgcn_wmma_f32_16x16x32_bf16(
            false, qf[c].v, false, kf[t][c].v, (short)0, s[t], false, false);

    // ---- online softmax (C layout: lane -> col l15, VGPR r -> row half*8+r)
#pragma unroll
    for (int r = 0; r < 8; ++r) {
      const int qrow = q0 + half * 8 + r;
      float sv0, sv1;
      {
        int key = kt + l15;
        bool ok = (key <= qrow) && (key >= qrow - (WIN_ - 1));
        sv0 = ok ? s[0][r] * scl : NEG_INF;
      }
      {
        int key = kt + 16 + l15;
        bool ok = (key <= qrow) && (key >= qrow - (WIN_ - 1));
        sv1 = ok ? s[1][r] * scl : NEG_INF;
      }
      float mx = fmaxf(mrow[r], fmaxf(sv0, sv1));
#pragma unroll
      for (int xm = 1; xm < 16; xm <<= 1) mx = fmaxf(mx, __shfl_xor(mx, xm, 32));
      const float alpha = __expf(mrow[r] - mx);
      float p0 = (sv0 <= -1.0e38f) ? 0.0f : __expf(sv0 - mx);
      float p1 = (sv1 <= -1.0e38f) ? 0.0f : __expf(sv1 - mx);
      float ps = p0 + p1;
#pragma unroll
      for (int xm = 1; xm < 16; xm <<= 1) ps += __shfl_xor(ps, xm, 32);
      lrow[r] = lrow[r] * alpha + ps;
      mrow[r] = mx;
#pragma unroll
      for (int d8 = 0; d8 < 8; ++d8) o[d8][r] *= alpha;
      const int row = half * 8 + r;
      P[row * 32 + l15]      = (__bf16)p0;
      P[row * 32 + 16 + l15] = (__bf16)p1;
    }
    asm volatile("s_wait_dscnt 0" ::: "memory");
    // ---- P (16x32) reloaded in A-fragment layout from LDS
    FragAB pf;
    {
      const int off = half ? 8 : 0;
      pf.h[0] = *(const v8bf*)&P[l15 * 32 + off];
      pf.h[1] = *(const v8bf*)&P[l15 * 32 + off + 16];
    }
    // ---- O += P * V (V fragments already resident)
#pragma unroll
    for (int d8 = 0; d8 < 8; ++d8)
      o[d8] = __builtin_amdgcn_wmma_f32_16x16x32_bf16(
          false, pf.v, false, vfr[d8].v, (short)0, o[d8], false, false);
  }
  // ---- finalize and store bf16 [L][H*HD]
#pragma unroll
  for (int d8 = 0; d8 < 8; ++d8)
#pragma unroll
    for (int r = 0; r < 8; ++r) {
      float val = o[d8][r] / lrow[r];
      AO[(size_t)(q0 + half * 8 + r) * (H_ * HD_) + h * HD_ + d8 * 16 + l15] =
          (__bf16)val;
    }
}

// ---------------------------------------------------------------- launcher

extern "C" void kernel_launch(void* const* d_in, const int* in_sizes, int n_in,
                              void* d_out, int out_size, void* d_ws, size_t ws_size,
                              hipStream_t stream) {
  const float* x  = (const float*)d_in[0];
  const int*  pos = (const int*)d_in[1];
  const float* Wq = (const float*)d_in[2];
  const float* Wk = (const float*)d_in[3];
  const float* Wv = (const float*)d_in[4];
  const float* Wo = (const float*)d_in[5];
  const float* qs = (const float*)d_in[6];
  const float* ks = (const float*)d_in[7];
  float* out = (float*)d_out;

  char* ws = (char*)d_ws;
  size_t off = 0;
  auto take = [&](size_t bytes) -> char* {
    char* p = ws + off;
    off += (bytes + 255) & ~(size_t)255;
    return p;
  };

  __bf16* Xb   = (__bf16*)take((size_t)L_ * D_ * 2);
  __bf16* WqT  = (__bf16*)take((size_t)(H_ * HD_) * D_ * 2);
  __bf16* WkT  = (__bf16*)take((size_t)(KV_ * HD_) * D_ * 2);
  __bf16* WvT  = (__bf16*)take((size_t)(KV_ * HD_) * D_ * 2);
  __bf16* WoT  = (__bf16*)take((size_t)D_ * (H_ * HD_) * 2);
  float*  qf   = (float*)take((size_t)L_ * H_ * HD_ * 4);
  float*  kf   = (float*)take((size_t)L_ * KV_ * HD_ * 4);
  float*  vf   = (float*)take((size_t)L_ * KV_ * HD_ * 4);
  __bf16* Qb   = (__bf16*)take((size_t)L_ * H_ * HD_ * 2);
  __bf16* Kb   = (__bf16*)take((size_t)L_ * KV_ * HD_ * 2);
  __bf16* VTb  = (__bf16*)take((size_t)L_ * KV_ * HD_ * 2);
  __bf16* AO   = (__bf16*)take((size_t)L_ * H_ * HD_ * 2);

  // 1) convert x, transpose-convert weights to bf16 [N][K]
  {
    int n = L_ * D_;
    f32_to_bf16_kernel<<<(n + 255) / 256, 256, 0, stream>>>(x, Xb, n);
  }
  {
    int n = D_ * (H_ * HD_);
    transpose_f32_to_bf16_kernel<<<(n + 255) / 256, 256, 0, stream>>>(Wq, WqT, D_, H_ * HD_);
  }
  {
    int n = D_ * (KV_ * HD_);
    transpose_f32_to_bf16_kernel<<<(n + 255) / 256, 256, 0, stream>>>(Wk, WkT, D_, KV_ * HD_);
    transpose_f32_to_bf16_kernel<<<(n + 255) / 256, 256, 0, stream>>>(Wv, WvT, D_, KV_ * HD_);
  }
  {
    int n = (H_ * HD_) * D_;
    transpose_f32_to_bf16_kernel<<<(n + 255) / 256, 256, 0, stream>>>(Wo, WoT, H_ * HD_, D_);
  }

  // 2) QKV projections (bf16 WMMA, fp32 out); K(=depth) is 2048 for all GEMMs
  gemm_bf16t_kernel<D_><<<dim3((H_ * HD_) / 64, L_ / 64), 128, 0, stream>>>(Xb, WqT, qf, L_, H_ * HD_);
  gemm_bf16t_kernel<D_><<<dim3((KV_ * HD_) / 64, L_ / 64), 128, 0, stream>>>(Xb, WkT, kf, L_, KV_ * HD_);
  gemm_bf16t_kernel<D_><<<dim3((KV_ * HD_) / 64, L_ / 64), 128, 0, stream>>>(Xb, WvT, vf, L_, KV_ * HD_);

  // 3) RMSNorm + RoPE -> head-major bf16; V -> [KV][HD][L] bf16
  rmsnorm_rope_kernel<<<dim3(L_, H_), 128, 0, stream>>>(qf, qs, pos, Qb, H_);
  rmsnorm_rope_kernel<<<dim3(L_, KV_), 128, 0, stream>>>(kf, ks, pos, Kb, KV_);
  {
    int n = L_ * KV_ * HD_;
    v_transpose_bf16_kernel<<<(n + 255) / 256, 256, 0, stream>>>(vf, VTb);
  }

  // 4) sliding-window flash attention
  attn_kernel<<<dim3(L_ / 64, H_), 128, 0, stream>>>(Qb, Kb, VTb, AO);

  // 5) output projection -> fp32 d_out
  gemm_bf16t_kernel<H_ * HD_><<<dim3(D_ / 64, L_ / 64), 128, 0, stream>>>(AO, WoT, out, L_, D_);
}
